// MLAKernelWrapper_34325378630012
// MI455X (gfx1250) — matrix operand order repaired and checked by reading, not compiled, not run on hardware
//
#include <hip/hip_runtime.h>

typedef __attribute__((ext_vector_type(16))) __bf16 v16bf;
typedef __attribute__((ext_vector_type(8)))  float  v8f;

union FragU {
  v16bf v;
  uint4 q[2];
  unsigned short u[16];
};

// Hardware bf16 convert: let clang lower fptrunc (native on gfx1250)
__device__ __forceinline__ unsigned short f2bf(float f) {
  __bf16 h = (__bf16)f;
  return __builtin_bit_cast(unsigned short, h);
}

__device__ __forceinline__ void load8f(const float* p, unsigned short* d) {
  float4 a = *(const float4*)p;
  float4 b = *(const float4*)(p + 4);
  d[0] = f2bf(a.x); d[1] = f2bf(a.y); d[2] = f2bf(a.z); d[3] = f2bf(a.w);
  d[4] = f2bf(b.x); d[5] = f2bf(b.y); d[6] = f2bf(b.z); d[7] = f2bf(b.w);
}

__device__ __forceinline__ v8f wmma_bf16(v16bf a, v16bf b, v8f c) {
  return __builtin_amdgcn_wmma_f32_16x16x32_bf16(false, a, false, b, (short)0, c, false, false);
}

// Async global -> LDS (CDNA5), 16B per lane. VDST = LDS byte offset (per-lane VGPR),
// VADDR = 64-bit global address (VGPR pair), saddr = off.  Tracked by ASYNCcnt.
__device__ __forceinline__ void async_b128(unsigned lds_off, const float* g) {
  asm volatile("global_load_async_to_lds_b128 %0, %1, off"
               :: "v"(lds_off), "v"((unsigned long long)(uintptr_t)g)
               : "memory");
}
__device__ __forceinline__ void wait_async0() {
  asm volatile("s_wait_asynccnt 0x0" ::: "memory");
}

#define NEG_BIG (-1e30f)

// ---------------------------------------------------------------------------
// Kernel 0: fused_q[b,h,l] = sum_d q_nope[b,h,d] * latent_to_k[l,h,d]  (bf16 out)
// Per head: M=32 (batch), N=512 (latent), K=128 (d). 128 blocks x 256 threads.
// ---------------------------------------------------------------------------
__global__ void k_fuse_q(const float* __restrict__ qn, const float* __restrict__ ltk,
                         unsigned short* __restrict__ fq) {
  const int h = blockIdx.x;
  const int tid = threadIdx.x, lane = tid & 31, w = tid >> 5;
  const int ln = lane & 15, half = lane >> 4;
  const int kbase = half * 8, koff = half * 16;
  const int mi = w >> 2, ng = w & 3;

  FragU a[4];
  const float* ap = qn + ((size_t)(16 * mi + ln) * 128 + h) * 128;
#pragma unroll
  for (int t = 0; t < 4; ++t) {
    load8f(ap + 32 * t + kbase, a[t].u);
    load8f(ap + 32 * t + kbase + 16, a[t].u + 8);
  }
#pragma unroll
  for (int j = 0; j < 8; ++j) {
    const int n = ng * 8 + j;
    const float* bp = ltk + ((size_t)(16 * n + ln) * 128 + h) * 128;
    v8f c = {};
#pragma unroll
    for (int t = 0; t < 4; ++t) {
      FragU bf;
      load8f(bp + 32 * t + kbase, bf.u);
      load8f(bp + 32 * t + kbase + 16, bf.u + 8);
      c = wmma_bf16(a[t].v, bf.v, c);
    }
#pragma unroll
    for (int r = 0; r < 8; ++r)
      fq[((size_t)(16 * mi + 8 * half + r) * 128 + h) * 512 + 16 * n + ln] = f2bf(c[r]);
  }
}

// ---------------------------------------------------------------------------
// Kernel 0b: q_pe fp32 -> bf16
// ---------------------------------------------------------------------------
__global__ void k_cvt_qpe(const float* __restrict__ src, unsigned short* __restrict__ dst) {
  int i = (blockIdx.x * 256 + threadIdx.x) * 4;
  dst[i + 0] = f2bf(src[i + 0]);
  dst[i + 1] = f2bf(src[i + 1]);
  dst[i + 2] = f2bf(src[i + 2]);
  dst[i + 3] = f2bf(src[i + 3]);
}

// ---------------------------------------------------------------------------
// Kernel 1: flash decode, split-K.  grid = B*8, block = 256 (8 waves).
// Wave w handles heads [16w,16w+16). 32-position kv tiles staged via
// GLOBAL_LOAD_ASYNC_TO_LDS (next tile's fetch overlaps current compute).
// ---------------------------------------------------------------------------
#define SROW 528   // kvS row stride (ushorts), 32 rows
#define TROW 40    // kvT row stride, 512 rows
#define KROW 72    // k_pe row stride, 32 rows
#define PROW 40    // P stage row stride, 16 rows per wave

// dynamic LDS layout (bytes):
#define OFF_KVRAW 0        // 32*512*4  = 65536 (fp32, async dest)
#define OFF_KPRAW 65536    // 32*64*4   = 8192  (fp32, async dest)
#define OFF_KVS   73728    // 32*528*2  = 33792
#define OFF_KVT   107520   // 512*40*2  = 40960
#define OFF_KPS   148480   // 32*72*2   = 4608
#define OFF_PST   153088   // 8*16*40*2 = 10240
#define SMEM_BYTES 163328

__global__ void __launch_bounds__(256)
k_flash(const float* __restrict__ kv, const float* __restrict__ kpe,
        const unsigned short* __restrict__ fq, const unsigned short* __restrict__ qpeB,
        const float* __restrict__ scale_p, const int* __restrict__ kvlen_p,
        float* __restrict__ Mp, float* __restrict__ Lp, float* __restrict__ Op) {
  extern __shared__ char smem[];
  float* kvRaw = (float*)(smem + OFF_KVRAW);
  float* kpRaw = (float*)(smem + OFF_KPRAW);
  unsigned short* kvS = (unsigned short*)(smem + OFF_KVS);
  unsigned short* kvT = (unsigned short*)(smem + OFF_KVT);
  unsigned short* kpS = (unsigned short*)(smem + OFF_KPS);
  unsigned short* pstAll = (unsigned short*)(smem + OFF_PST);

  const int b = blockIdx.x >> 3;
  const int sp = blockIdx.x & 7;
  const int tid = threadIdx.x, lane = tid & 31, w = tid >> 5;
  const int ln = lane & 15, half = lane >> 4;
  const int kbase = half * 8, koff = half * 16;
  unsigned short* pst = pstAll + w * (16 * PROW);

  const int hbase = 16 * w;
  const int h = hbase + ln;
  const float sc = *scale_p;
  const int kvlen = kvlen_p[b];
  const int chunk0 = sp * 512;

  // LDS byte offsets for async destinations (flat LDS addr low 32 bits == offset)
  const unsigned kvraw_base = (unsigned)(uintptr_t)(void*)kvRaw;
  const unsigned kpraw_base = (unsigned)(uintptr_t)(void*)kpRaw;

  // Preload A fragments: fused_q (16x512 over K=latent) and q_pe (16x64)
  FragU aq[16], ape[2];
  {
    const unsigned short* fp = fq + ((size_t)b * 128 + h) * 512;
#pragma unroll
    for (int t = 0; t < 16; ++t) {
      aq[t].q[0] = *(const uint4*)(fp + 32 * t + kbase);
      aq[t].q[1] = *(const uint4*)(fp + 32 * t + kbase + 16);
    }
    const unsigned short* pp = qpeB + ((size_t)b * 128 + h) * 64;
#pragma unroll
    for (int t = 0; t < 2; ++t) {
      ape[t].q[0] = *(const uint4*)(pp + 32 * t + kbase);
      ape[t].q[1] = *(const uint4*)(pp + 32 * t + kbase + 16);
    }
  }

  v8f zero = {};
  v8f oacc[32];
#pragma unroll
  for (int n = 0; n < 32; ++n) oacc[n] = zero;
  float mrow[8], lrow[8];
#pragma unroll
  for (int r = 0; r < 8; ++r) { mrow[r] = NEG_BIG; lrow[r] = 0.f; }

  if (chunk0 <= kvlen) {
    // prologue: async-fetch tile 0 (contiguous 32 rows x 512 / 64 fp32)
    {
      const float* gkv = kv + ((size_t)b * 4096 + chunk0) * 512;
      const float* gkp = kpe + ((size_t)b * 4096 + chunk0) * 64;
#pragma unroll
      for (int j = 0; j < 16; ++j) {
        unsigned c = tid + 256 * j;
        async_b128(kvraw_base + c * 16, gkv + c * 4);
      }
#pragma unroll
      for (int j = 0; j < 2; ++j) {
        unsigned c = tid + 256 * j;
        async_b128(kpraw_base + c * 16, gkp + c * 4);
      }
    }

    for (int it = 0; it < 16; ++it) {
      const int pbase = chunk0 + 32 * it;
      wait_async0();       // this wave's async tile data landed in LDS
      __syncthreads();     // all waves' data landed; prior compute reads done

      // convert raw fp32 tile -> bf16 kvS (row-major) + kvT (transposed) + kpS
      {
        const int row = tid >> 3;
        const int c0 = (tid & 7) * 64;
        const float* src = kvRaw + row * 512 + c0;
        unsigned short* dS = kvS + row * SROW + c0;
#pragma unroll
        for (int i = 0; i < 64; i += 4) {
          float4 x = *(const float4*)(src + i);
          unsigned short u0 = f2bf(x.x), u1 = f2bf(x.y), u2 = f2bf(x.z), u3 = f2bf(x.w);
          dS[i + 0] = u0; dS[i + 1] = u1; dS[i + 2] = u2; dS[i + 3] = u3;
          kvT[(c0 + i + 0) * TROW + row] = u0;
          kvT[(c0 + i + 1) * TROW + row] = u1;
          kvT[(c0 + i + 2) * TROW + row] = u2;
          kvT[(c0 + i + 3) * TROW + row] = u3;
        }
        const int c1 = (tid & 7) * 8;
        const float* s2 = kpRaw + row * 64 + c1;
        float4 y0 = *(const float4*)s2;
        float4 y1 = *(const float4*)(s2 + 4);
        unsigned short* dK = kpS + row * KROW + c1;
        dK[0] = f2bf(y0.x); dK[1] = f2bf(y0.y); dK[2] = f2bf(y0.z); dK[3] = f2bf(y0.w);
        dK[4] = f2bf(y1.x); dK[5] = f2bf(y1.y); dK[6] = f2bf(y1.z); dK[7] = f2bf(y1.w);
      }
      __syncthreads();     // conversion complete -> raw buffer reusable

      // overlap: async-fetch next tile into raw buffer while we compute
      if (it < 15) {
        const int nbase = chunk0 + 32 * (it + 1);
        const float* gkv = kv + ((size_t)b * 4096 + nbase) * 512;
        const float* gkp = kpe + ((size_t)b * 4096 + nbase) * 64;
#pragma unroll
        for (int j = 0; j < 16; ++j) {
          unsigned c = tid + 256 * j;
          async_b128(kvraw_base + c * 16, gkv + c * 4);
        }
#pragma unroll
        for (int j = 0; j < 2; ++j) {
          unsigned c = tid + 256 * j;
          async_b128(kpraw_base + c * 16, gkp + c * 4);
        }
      }

      // scores for two 16-position sub-tiles
      v8f s01[2];
#pragma unroll
      for (int sub = 0; sub < 2; ++sub) {
        v8f s = zero;
        const unsigned short* rowp = kvS + (sub * 16 + ln) * SROW;
#pragma unroll
        for (int t = 0; t < 16; ++t) {
          FragU bf;
          bf.q[0] = *(const uint4*)(rowp + 32 * t + kbase);
          bf.q[1] = *(const uint4*)(rowp + 32 * t + kbase + 16);
          s = wmma_bf16(aq[t].v, bf.v, s);
        }
        const unsigned short* rp = kpS + (sub * 16 + ln) * KROW;
#pragma unroll
        for (int t = 0; t < 2; ++t) {
          FragU bf;
          bf.q[0] = *(const uint4*)(rp + 32 * t + kbase);
          bf.q[1] = *(const uint4*)(rp + 32 * t + kbase + 16);
          s = wmma_bf16(ape[t].v, bf.v, s);
        }
        const int kg = pbase + sub * 16 + ln;
        const bool valid = (kg <= kvlen);
#pragma unroll
        for (int r = 0; r < 8; ++r) s[r] = valid ? s[r] * sc : NEG_BIG;
        s01[sub] = s;
      }

      // online softmax (rows live in 16-lane halves, 8 rows per lane via VGPR idx)
      float rm[8];
#pragma unroll
      for (int r = 0; r < 8; ++r) rm[r] = fmaxf(s01[0][r], s01[1][r]);
#pragma unroll
      for (int m = 1; m < 16; m <<= 1)
#pragma unroll
        for (int r = 0; r < 8; ++r) rm[r] = fmaxf(rm[r], __shfl_xor(rm[r], m));
      float corr[8];
#pragma unroll
      for (int r = 0; r < 8; ++r) {
        float mn = fmaxf(mrow[r], rm[r]);
        corr[r] = __expf(mrow[r] - mn);
        mrow[r] = mn;
      }
      float rs[8];
#pragma unroll
      for (int r = 0; r < 8; ++r) {
        float p0 = __expf(s01[0][r] - mrow[r]);
        float p1 = __expf(s01[1][r] - mrow[r]);
        pst[(8 * half + r) * PROW + ln] = f2bf(p0);        // C layout -> LDS stage
        pst[(8 * half + r) * PROW + 16 + ln] = f2bf(p1);
        rs[r] = p0 + p1;
      }
#pragma unroll
      for (int m = 1; m < 16; m <<= 1)
#pragma unroll
        for (int r = 0; r < 8; ++r) rs[r] += __shfl_xor(rs[r], m);
#pragma unroll
      for (int r = 0; r < 8; ++r) lrow[r] = lrow[r] * corr[r] + rs[r];
#pragma unroll
      for (int n = 0; n < 32; ++n)
#pragma unroll
        for (int r = 0; r < 8; ++r) oacc[n][r] *= corr[r];

      // PV: O[16,512] += P(16x32) x kv(32x512)
      FragU pf;
      pf.q[0] = *(const uint4*)(pst + ln * PROW + kbase);
      pf.q[1] = *(const uint4*)(pst + ln * PROW + kbase + 16);
#pragma unroll
      for (int n = 0; n < 32; ++n) {
        FragU bf;
        const unsigned short* tp = kvT + (16 * n + ln) * TROW;
        bf.q[0] = *(const uint4*)(tp + kbase);
        bf.q[1] = *(const uint4*)(tp + kbase + 16);
        oacc[n] = wmma_bf16(pf.v, bf.v, oacc[n]);
      }
    }
  }

  // store split partials
  const size_t pb = ((size_t)b * 8 + sp) * 128;
#pragma unroll
  for (int n = 0; n < 32; ++n)
#pragma unroll
    for (int r = 0; r < 8; ++r)
      Op[(pb + hbase + 8 * half + r) * 512 + 16 * n + ln] = oacc[n][r];
  if (ln == 0) {
#pragma unroll
    for (int r = 0; r < 8; ++r) {
      Mp[pb + hbase + 8 * half + r] = mrow[r];
      Lp[pb + hbase + 8 * half + r] = lrow[r];
    }
  }
}

// ---------------------------------------------------------------------------
// Kernel 2: reduce the 8 splits -> latent_out bf16 [B][H][512]
// ---------------------------------------------------------------------------
__global__ void k_reduce(const float* __restrict__ Mp, const float* __restrict__ Lp,
                         const float* __restrict__ Op, unsigned short* __restrict__ latB) {
  const int b = blockIdx.x >> 7;
  const int h = blockIdx.x & 127;
  float ms[8], ls[8];
  float mg = NEG_BIG;
#pragma unroll
  for (int s = 0; s < 8; ++s) {
    ms[s] = Mp[((size_t)b * 8 + s) * 128 + h];
    ls[s] = Lp[((size_t)b * 8 + s) * 128 + h];
    mg = fmaxf(mg, ms[s]);
  }
  float wt[8];
  float lg = 0.f;
#pragma unroll
  for (int s = 0; s < 8; ++s) { wt[s] = __expf(ms[s] - mg); lg += ls[s] * wt[s]; }
  const float inv = 1.f / lg;
  for (int l = threadIdx.x; l < 512; l += 256) {
    float acc = 0.f;
#pragma unroll
    for (int s = 0; s < 8; ++s)
      acc += wt[s] * Op[(((size_t)b * 8 + s) * 128 + h) * 512 + l];
    latB[((size_t)b * 128 + h) * 512 + l] = f2bf(acc * inv);
  }
}

// ---------------------------------------------------------------------------
// Kernel 3: out[b, h*128+v] = sum_l latent_out[b,h,l] * latent_to_v[l,h,v]
// Per head: M=32 (batch), N=128 (v), K=512 (latent). 128 blocks x 256 threads.
// ---------------------------------------------------------------------------
__global__ void k_proj(const unsigned short* __restrict__ latB, const float* __restrict__ ltv,
                       float* __restrict__ out) {
  const int h = blockIdx.x;
  const int tid = threadIdx.x, lane = tid & 31, w = tid >> 5;
  const int ln = lane & 15, half = lane >> 4;
  const int kbase = half * 8, koff = half * 16;
  const int mi = w >> 2;

  FragU a[16];
  const unsigned short* apb = latB + ((size_t)(16 * mi + ln) * 128 + h) * 512;
#pragma unroll
  for (int t = 0; t < 16; ++t) {
    a[t].q[0] = *(const uint4*)(apb + 32 * t + kbase);
    a[t].q[1] = *(const uint4*)(apb + 32 * t + kbase + 16);
  }
#pragma unroll
  for (int nn = 0; nn < 2; ++nn) {
    const int n = (w & 3) * 2 + nn;
    v8f c = {};
#pragma unroll
    for (int t = 0; t < 16; ++t) {
      FragU bf;
#pragma unroll
      for (int e = 0; e < 16; ++e)
        bf.u[e] = f2bf(ltv[(size_t)(32 * t + kbase + (e & 7) + (e >> 3) * 16) * 16384 + h * 128 + 16 * n + ln]);
      c = wmma_bf16(a[t].v, bf.v, c);
    }
#pragma unroll
    for (int r = 0; r < 8; ++r)
      out[(size_t)(16 * mi + 8 * half + r) * 16384 + h * 128 + 16 * n + ln] = c[r];
  }
}

// ---------------------------------------------------------------------------
extern "C" void kernel_launch(void* const* d_in, const int* in_sizes, int n_in,
                              void* d_out, int out_size, void* d_ws, size_t ws_size,
                              hipStream_t stream) {
  const float* qn = (const float*)d_in[0];      // [32,128,1,128]
  const float* qpe = (const float*)d_in[1];     // [32,128,1,64]
  const float* kv = (const float*)d_in[2];      // [32,4096,512]
  const float* kpe = (const float*)d_in[3];     // [32,4096,64]
  const float* ltk = (const float*)d_in[4];     // [512,128,128]
  const float* ltv = (const float*)d_in[5];     // [512,128,128]
  const float* scale_p = (const float*)d_in[6]; // [1]
  const int* kvlen = (const int*)d_in[7];       // [32]
  float* out = (float*)d_out;                   // [32,16384]

  unsigned short* fq = (unsigned short*)d_ws;                      // 32*128*512 bf16
  unsigned short* qpeB = fq + (size_t)32 * 128 * 512;              // 32*128*64  bf16
  float* Mp = (float*)(qpeB + (size_t)32 * 128 * 64);              // 32*8*128
  float* Lp = Mp + 32 * 8 * 128;                                   // 32*8*128
  float* Op = Lp + 32 * 8 * 128;                                   // 32*8*128*512 f32
  unsigned short* latB = (unsigned short*)(Op + (size_t)32 * 8 * 128 * 512);

  k_fuse_q<<<128, 256, 0, stream>>>(qn, ltk, fq);
  k_cvt_qpe<<<256, 256, 0, stream>>>(qpe, qpeB);
  k_flash<<<256, 256, SMEM_BYTES, stream>>>(kv, kpe, fq, qpeB, scale_p, kvlen, Mp, Lp, Op);
  k_reduce<<<4096, 256, 0, stream>>>(Mp, Lp, Op, latB);
  k_proj<<<128, 256, 0, stream>>>(latB, ltv, out);
}
